// AttnProcessor2_0_NS2_62809601737285
// MI455X (gfx1250) — matrix operand order repaired and checked by reading, not compiled, not run on hardware
//
#include <hip/hip_runtime.h>
#include <hip/hip_bf16.h>

typedef __attribute__((ext_vector_type(16))) __bf16 v16bf;
typedef __attribute__((ext_vector_type(8)))  __bf16 v8bf;
typedef __attribute__((ext_vector_type(8)))  float  v8f;
typedef __attribute__((ext_vector_type(4)))  int    v4i;

#define BB 2
#define SS 4096
#define CC 640
#define HH 10
#define DH 64
#define MM (BB * SS)   /* 8192 rows */
#define ATT_SCALE 0.125f

struct U32x8 { uint4 a, b; };

static __device__ __forceinline__ v16bf cat16(uint4 lo, uint4 hi) {
  U32x8 t; t.a = lo; t.b = hi;
  return __builtin_bit_cast(v16bf, t);
}

static __device__ __forceinline__ v8f wmma_bf16(v16bf a, v16bf b, v8f c) {
  // (neg_a, A, neg_b, B, c_mod, C, reuse_a, reuse_b)
  return __builtin_amdgcn_wmma_f32_16x16x32_bf16(false, a, false, b, (short)0, c,
                                                 false, false);
}

// 16-byte global -> LDS copy. Uses the gfx1250 async-DMA path (ASYNCcnt) when
// the builtin exists; otherwise a plain VGPR round-trip.
static __device__ __forceinline__ void async_copy16(__bf16* lds_dst,
                                                    const __bf16* gsrc) {
#if __has_builtin(__builtin_amdgcn_global_load_async_to_lds_b128)
  __builtin_amdgcn_global_load_async_to_lds_b128(
      (__attribute__((address_space(1))) v4i*)gsrc,
      (__attribute__((address_space(3))) v4i*)lds_dst, 0, 0);
#else
  *(uint4*)lds_dst = *(const uint4*)gsrc;
#endif
}

// ---------------------------------------------------------------------------
// Kernel 1: Q/K/V projection.  Y = X @ W^T, f32 inputs -> bf16 outputs.
// grid = (M/128, C/64, 3), block = 256 (8 waves). Each wave owns a 16x64 tile.
// ---------------------------------------------------------------------------
__global__ __launch_bounds__(256) void qkv_proj_kernel(
    const float* __restrict__ X,
    const float* __restrict__ Wq, const float* __restrict__ Wk,
    const float* __restrict__ Wv,
    __hip_bfloat16* __restrict__ Qo, __hip_bfloat16* __restrict__ Ko,
    __hip_bfloat16* __restrict__ Vo) {
  __shared__ __bf16 xs[128][40];  // 128x32 bf16 tile, 80B row stride (16B aligned)
  __shared__ __bf16 ws[64][40];

  const int tid  = threadIdx.x;
  const int wave = tid >> 5, lane = tid & 31;
  const int lm = lane & 15, half = lane >> 4;
  const int m0 = blockIdx.x * 128;
  const int n0 = blockIdx.y * 64;
  const float* W = (blockIdx.z == 0) ? Wq : (blockIdx.z == 1) ? Wk : Wv;
  __bf16* Y = reinterpret_cast<__bf16*>((blockIdx.z == 0)   ? Qo
                                        : (blockIdx.z == 1) ? Ko
                                                            : Vo);

  v8f acc[4] = {};

  const int xrow = tid >> 1, xcb = (tid & 1) * 16;  // X loader: 128 rows x 32
  const int wrow = tid >> 2, wcb = (tid & 3) * 8;   // W loader: 64 rows x 32

  for (int kk = 0; kk < CC; kk += 32) {
    const float* xsrc = X + (size_t)(m0 + xrow) * CC + kk + xcb;
#pragma unroll
    for (int j = 0; j < 16; ++j) xs[xrow][xcb + j] = (__bf16)xsrc[j];
    const float* wsrc = W + (size_t)(n0 + wrow) * CC + kk + wcb;
#pragma unroll
    for (int j = 0; j < 8; ++j) ws[wrow][wcb + j] = (__bf16)wsrc[j];
    if (kk + 32 < CC) {  // prefetch next K-slab (global_prefetch_b8)
      __builtin_prefetch(xsrc + 32, 0, 0);
      __builtin_prefetch(wsrc + 32, 0, 0);
    }
    __syncthreads();

    // A fragment: lane = row, K in two 8-runs per half-wave
    const uint4* a0 = (const uint4*)&xs[wave * 16 + lm][half * 8];
    const uint4* a1 = (const uint4*)&xs[wave * 16 + lm][16 + half * 8];
    v16bf afrag = cat16(a0[0], a1[0]);
#pragma unroll
    for (int nb = 0; nb < 4; ++nb) {
      // B fragment: lane = output column n, K in two 8-runs per half-wave
      const uint4* bp = (const uint4*)&ws[nb * 16 + lm][half * 8];
      acc[nb] = wmma_bf16(afrag, cat16(bp[0], bp[2]), acc[nb]);
    }
    __syncthreads();
  }

#pragma unroll
  for (int nb = 0; nb < 4; ++nb)
#pragma unroll
    for (int r = 0; r < 8; ++r) {
      const int row = m0 + wave * 16 + r + half * 8;
      const int col = n0 + nb * 16 + lm;
      Y[(size_t)row * CC + col] = (__bf16)acc[nb][r];
    }
}

// ---------------------------------------------------------------------------
// Kernel 2: flash attention over bf16 Q/K/V.  grid = (S/128, H, B), block=256.
// Each wave handles 16 query rows; 32-key tiles streamed through ping-pong
// LDS buffers. K tiles arrive via async-to-LDS DMA issued one iteration
// ahead (spanning the whole compute phase); one barrier per iteration.
// ---------------------------------------------------------------------------
__global__ __launch_bounds__(256) void flash_attn_kernel(
    const __hip_bfloat16* __restrict__ Qi, const __hip_bfloat16* __restrict__ Ki,
    const __hip_bfloat16* __restrict__ Vi, __hip_bfloat16* __restrict__ Oo) {
  __shared__ __bf16 ks_[2][32][72];    // [buf][key][d]   32x64, 144B stride
  __shared__ __bf16 vs_[2][64][40];    // [buf][d][key]   transposed, 80B stride
  __shared__ __bf16 ps_[8][16][40];    // per-wave P scratch, 80B stride

  const __bf16* Q = reinterpret_cast<const __bf16*>(Qi);
  const __bf16* K = reinterpret_cast<const __bf16*>(Ki);
  const __bf16* V = reinterpret_cast<const __bf16*>(Vi);
  __bf16* O = reinterpret_cast<__bf16*>(Oo);

  const int tid  = threadIdx.x;
  const int wave = tid >> 5, lane = tid & 31;
  const int lm = lane & 15, half = lane >> 4;
  const int h = blockIdx.y, b = blockIdx.z;
  const int q0 = blockIdx.x * 128 + wave * 16;
  const size_t base = ((size_t)b * SS) * CC + (size_t)h * DH;

  // Preload this wave's Q fragments (16 rows x Dh=64 -> two K=32 A-fragments)
  const __bf16* qrow = Q + base + (size_t)(q0 + lm) * CC;
  v16bf aq[2];
#pragma unroll
  for (int kstep = 0; kstep < 2; ++kstep) {
    const uint4* lo = (const uint4*)(qrow + kstep * 32 + half * 8);
    const uint4* hi = (const uint4*)(qrow + kstep * 32 + 16 + half * 8);
    aq[kstep] = cat16(lo[0], hi[0]);
  }

  v8f oacc[4] = {};
  float mstat[8], lstat[8];
#pragma unroll
  for (int r = 0; r < 8; ++r) { mstat[r] = -1e30f; lstat[r] = 0.f; }

  const int lkey = tid >> 3, ldb = (tid & 7) * 8;  // tile loader coords
  const __bf16* kg = K + base + (size_t)lkey * CC + ldb;
  const __bf16* vg = V + base + (size_t)lkey * CC + ldb;
  const size_t tile_step = (size_t)32 * CC;

  // Prologue: tile 0 -> buffer 0 (K via async DMA, V into registers)
  async_copy16(&ks_[0][lkey][ldb], kg);
  uint4 vvec = *(const uint4*)vg;

  const int NIT = SS / 32;  // 128
  for (int it = 0; it < NIT; ++it) {
    const int cur = it & 1;

    // Transpose-store the prefetched V tile into vs_[cur]
    {
      v8bf vv = __builtin_bit_cast(v8bf, vvec);
#pragma unroll
      for (int j = 0; j < 8; ++j) vs_[cur][ldb + j][lkey] = vv[j];
    }
    // Ensure this wave's async K DMA for the current tile has landed, then
    // rendezvous so all waves' tile data is visible.
    asm volatile("s_wait_asynccnt 0x0" ::: "memory");
    __syncthreads();

    // Kick off next tile's loads into the other buffer; the async DMA runs
    // concurrently with the whole compute phase below.
    if (it + 1 < NIT) {
      async_copy16(&ks_[cur ^ 1][lkey][ldb], kg + (size_t)(it + 1) * tile_step);
      vvec = *(const uint4*)(vg + (size_t)(it + 1) * tile_step);
    }

    // S = Q @ K^T for this 16x32 score strip (two 16-key WMMA column tiles)
    v8f sc[2];
#pragma unroll
    for (int kt = 0; kt < 2; ++kt) {
      v8f c = {};
#pragma unroll
      for (int kstep = 0; kstep < 2; ++kstep) {
        const uint4* bp =
            (const uint4*)&ks_[cur][kt * 16 + lm][kstep * 32 + half * 8];
        c = wmma_bf16(aq[kstep], cat16(bp[0], bp[2]), c);
      }
      sc[kt] = c;
    }

    // Online softmax. C-layout: element r <-> row (r + 8*half); row spans the
    // 16 lanes of one half-wave, so xor-shuffles 1/2/4/8 reduce a row.
#pragma unroll
    for (int r = 0; r < 8; ++r) {
      float s0 = sc[0][r] * ATT_SCALE;
      float s1 = sc[1][r] * ATT_SCALE;
      float mx = fmaxf(s0, s1);
#pragma unroll
      for (int off = 1; off < 16; off <<= 1)
        mx = fmaxf(mx, __shfl_xor(mx, off, 32));
      float mnew = fmaxf(mstat[r], mx);
      float corr = __expf(mstat[r] - mnew);
      float p0 = __expf(s0 - mnew);
      float p1 = __expf(s1 - mnew);
      float rs = p0 + p1;
#pragma unroll
      for (int off = 1; off < 16; off <<= 1) rs += __shfl_xor(rs, off, 32);
      lstat[r] = lstat[r] * corr + rs;
      mstat[r] = mnew;
#pragma unroll
      for (int nb = 0; nb < 4; ++nb) oacc[nb][r] *= corr;
      ps_[wave][r + half * 8][lm]      = (__bf16)p0;
      ps_[wave][r + half * 8][16 + lm] = (__bf16)p1;
    }
    // wave-private LDS round-trip: ensure stores landed before A-frag reload
    asm volatile("s_wait_dscnt 0x0" ::: "memory");

    // O += P(16x32) @ V(32x64)
    const uint4* pa0 = (const uint4*)&ps_[wave][lm][half * 8];
    const uint4* pa1 = (const uint4*)&ps_[wave][lm][16 + half * 8];
    v16bf ap = cat16(pa0[0], pa1[0]);
#pragma unroll
    for (int nb = 0; nb < 4; ++nb) {
      const uint4* bp = (const uint4*)&vs_[cur][nb * 16 + lm][half * 8];
      oacc[nb] = wmma_bf16(ap, cat16(bp[0], bp[2]), oacc[nb]);
    }
    // No trailing barrier: the next iteration's pre-barrier writes only touch
    // the opposite buffer, and the single barrier per iteration forbids any
    // wave from being a full buffer-cycle (2 iterations) ahead.
  }

#pragma unroll
  for (int nb = 0; nb < 4; ++nb)
#pragma unroll
    for (int r = 0; r < 8; ++r) {
      const int row = q0 + r + half * 8;
      O[base + (size_t)row * CC + nb * 16 + lm] =
          (__bf16)(oacc[nb][r] / lstat[r]);
    }
}

// ---------------------------------------------------------------------------
// Kernel 3: output projection + bias + residual.  f32 output.
// ---------------------------------------------------------------------------
__global__ __launch_bounds__(256) void out_proj_kernel(
    const __hip_bfloat16* __restrict__ Ai, const float* __restrict__ Wo,
    const float* __restrict__ bias, const float* __restrict__ resid,
    float* __restrict__ Y) {
  __shared__ __bf16 xs[128][40];
  __shared__ __bf16 ws[64][40];
  const __bf16* A = reinterpret_cast<const __bf16*>(Ai);

  const int tid  = threadIdx.x;
  const int wave = tid >> 5, lane = tid & 31;
  const int lm = lane & 15, half = lane >> 4;
  const int m0 = blockIdx.x * 128;
  const int n0 = blockIdx.y * 64;

  v8f acc[4] = {};
  const int xrow = tid >> 1, xcb = (tid & 1) * 16;
  const int wrow = tid >> 2, wcb = (tid & 3) * 8;

  for (int kk = 0; kk < CC; kk += 32) {
    const uint4* asrc = (const uint4*)(A + (size_t)(m0 + xrow) * CC + kk + xcb);
    *(uint4*)&xs[xrow][xcb]     = asrc[0];
    *(uint4*)&xs[xrow][xcb + 8] = asrc[1];
    const float* wsrc = Wo + (size_t)(n0 + wrow) * CC + kk + wcb;
#pragma unroll
    for (int j = 0; j < 8; ++j) ws[wrow][wcb + j] = (__bf16)wsrc[j];
    if (kk + 32 < CC) {
      __builtin_prefetch((const void*)((const __bf16*)asrc + 32), 0, 0);
      __builtin_prefetch(wsrc + 32, 0, 0);
    }
    __syncthreads();

    const uint4* a0 = (const uint4*)&xs[wave * 16 + lm][half * 8];
    const uint4* a1 = (const uint4*)&xs[wave * 16 + lm][16 + half * 8];
    v16bf afrag = cat16(a0[0], a1[0]);
#pragma unroll
    for (int nb = 0; nb < 4; ++nb) {
      const uint4* bp = (const uint4*)&ws[nb * 16 + lm][half * 8];
      acc[nb] = wmma_bf16(afrag, cat16(bp[0], bp[2]), acc[nb]);
    }
    __syncthreads();
  }

#pragma unroll
  for (int nb = 0; nb < 4; ++nb)
#pragma unroll
    for (int r = 0; r < 8; ++r) {
      const int row = m0 + wave * 16 + r + half * 8;
      const int col = n0 + nb * 16 + lm;
      Y[(size_t)row * CC + col] =
          acc[nb][r] + bias[col] + resid[(size_t)row * CC + col];
    }
}

// ---------------------------------------------------------------------------
extern "C" void kernel_launch(void* const* d_in, const int* in_sizes, int n_in,
                              void* d_out, int out_size, void* d_ws,
                              size_t ws_size, hipStream_t stream) {
  const float* x  = (const float*)d_in[0];
  const float* Wq = (const float*)d_in[1];
  const float* Wk = (const float*)d_in[2];
  const float* Wv = (const float*)d_in[3];
  const float* Wo = (const float*)d_in[4];
  const float* bo = (const float*)d_in[5];
  float* out = (float*)d_out;

  const size_t elems = (size_t)MM * CC;  // 8192*640
  __hip_bfloat16* q  = (__hip_bfloat16*)d_ws;
  __hip_bfloat16* k  = q + elems;
  __hip_bfloat16* v  = k + elems;
  __hip_bfloat16* ao = v + elems;  // total ws: 4 * 10.5 MB = ~42 MB

  qkv_proj_kernel<<<dim3(MM / 128, CC / 64, 3), 256, 0, stream>>>(
      x, Wq, Wk, Wv, q, k, v);
  flash_attn_kernel<<<dim3(SS / 128, HH, BB), 256, 0, stream>>>(q, k, v, ao);
  out_proj_kernel<<<dim3(MM / 128, CC / 64), 256, 0, stream>>>(ao, Wo, bo, x,
                                                               out);
}
